// presetUResNet_52115133170192
// MI455X (gfx1250) — matrix operand order repaired and checked by reading, not compiled, not run on hardware
//
#include <hip/hip_runtime.h>

typedef __attribute__((ext_vector_type(16))) _Float16 v16h;
typedef __attribute__((ext_vector_type(8)))  _Float16 v8h;
typedef __attribute__((ext_vector_type(8)))  float    v8f;

// ---------------------------------------------------------------------------
// Input prep: x*occ -> f16 channel-last, padded to 16 channels (c=0 holds data)
// ---------------------------------------------------------------------------
__global__ void prep_in(const float* __restrict__ x, const float* __restrict__ occ,
                        _Float16* __restrict__ a, long n) {
  long i = (long)blockIdx.x * blockDim.x + threadIdx.x;
  if (i >= n) return;
  float v = x[i] * occ[i];
  _Float16* p = a + i * 16;
  p[0] = (_Float16)v;
#pragma unroll
  for (int c = 1; c < 16; ++c) p[c] = (_Float16)0.f;
}

// ---------------------------------------------------------------------------
// 2x2x2 max-pool of occupancy mask (f32 -> f32), D/H/W are OUTPUT dims
// ---------------------------------------------------------------------------
__global__ void pool2(const float* __restrict__ in, float* __restrict__ out,
                      int D, int H, int W) {
  long i = (long)blockIdx.x * blockDim.x + threadIdx.x;
  long n = (long)D * H * W;
  if (i >= n) return;
  int xx = (int)(i % W);
  int yy = (int)((i / W) % H);
  int zz = (int)(i / ((long)W * H));
  int Wi = 2 * W, Hi = 2 * H;
  float mx = 0.f;
#pragma unroll
  for (int dz = 0; dz < 2; ++dz)
#pragma unroll
    for (int dy = 0; dy < 2; ++dy)
#pragma unroll
      for (int dx = 0; dx < 2; ++dx)
        mx = fmaxf(mx, in[((long)(2 * zz + dz) * Hi + (2 * yy + dy)) * Wi + (2 * xx + dx)]);
  out[i] = mx;
}

// ---------------------------------------------------------------------------
// Weight swizzle: f32 OIDHW -> f16 B tiles laid out per-lane for
// v_wmma_f32_16x16x32_f16.  B_sw[nt][bv][r][lane][j], lane: n=lane%16, h=lane/16,
// K = h*16 + j, kk = r*32 + K flattens (tap, cin) with cin inner.
// mode 0: 27-tap conv3   mode 1: 8-tap down2   mode 2: up2 (bv = fine (z,y)
// parity, kk = xparity*Cin + c, taps spatially flipped per conv_transpose).
// ---------------------------------------------------------------------------
__global__ void swizzle_w(const float* __restrict__ w, _Float16* __restrict__ bsw,
                          int actCin, int wCin, int Cout, int T, int R, int Nt, int mode) {
  int BV = (mode == 2) ? 4 : 1;
  long total = (long)Nt * BV * R * 512;
  long idx = (long)blockIdx.x * blockDim.x + threadIdx.x;
  if (idx >= total) return;
  int j = (int)(idx & 15);
  int lane = (int)((idx >> 4) & 31);
  long rest = idx >> 9;
  int r = (int)(rest % R); rest /= R;
  int bv = (int)(rest % BV);
  int nt = (int)(rest / BV);
  int n = lane & 15, hh = lane >> 4;
  int kk = r * 32 + hh * 16 + j;
  int o = nt * 16 + n;
  float val = 0.f;
  if (mode == 2) {
    int xp = kk / actCin;
    int c = kk - xp * actCin;
    int vz = bv >> 1, vy = bv & 1;
    if (xp < 2 && c < wCin && o < Cout) {
      int t = (1 - vz) * 4 + (1 - vy) * 2 + (1 - xp); // conv_transpose tap flip
      val = w[((long)o * wCin + c) * 8 + t];
    }
  } else {
    int t = kk / actCin;
    int c = kk - t * actCin;
    if (t < T && c < wCin && o < Cout)
      val = w[((long)o * wCin + c) * T + t];
  }
  bsw[idx] = (_Float16)val;
}

// ---------------------------------------------------------------------------
// Unified implicit-GEMM masked conv using v_wmma_f32_16x16x32_f16.
// One wave handles (16-wide x tile, MROWS y rows, ALL NTILES outch tiles):
//   - A fragment loaded once, reused by NTILES wmmas
//   - B fragment loaded once, reused by MROWS wmmas
// WVS waves (consecutive z slices) per block share the B panel through LDS,
// staged in 8-chunk slabs (<=32KB) -> global B traffic / WVS, inner reads are
// ds_load_b128.  MODE 2 (B depends on z/y parity) keeps 1 wave + global B.
// Grid (tilesX, Hout/MROWS, Dout/WVS): decode is bit-ops, EXEC all-ones.
// ---------------------------------------------------------------------------
template <int MODE, int CIN, int R, int NTILES, int MROWS, int WVS>
__global__ __launch_bounds__(32 * WVS) void conv_wmma(
    const _Float16* __restrict__ in, const _Float16* __restrict__ bsw,
    const float* __restrict__ mask, _Float16* __restrict__ outh,
    float* __restrict__ outf,
    int Din, int Hin, int Win, int Dout, int Hout, int Wout,
    int inStride, int inOff, int outStride, int outOff,
    int relu, int coutValid, long outVol) {
  constexpr int CSH = (CIN == 16) ? 4 : (CIN == 32) ? 5 : 6;
  constexpr int BV = (MODE == 2) ? 4 : 1;
  constexpr bool LDSB = (MODE != 2);
  constexpr int RB = 8;  // chunks per LDS slab (shift-only decode relies on 8)
  __shared__ _Float16 lb[LDSB ? NTILES * RB * 512 : 32];

  int tid = threadIdx.x;
  int lane = tid & 31;
  int wave = tid >> 5;
  int z = blockIdx.z * WVS + wave;
  int y0 = blockIdx.y * MROWS;
  int xb = blockIdx.x << 4;
  int h = lane >> 4, m = lane & 15;
  int x = xb + m;  // A-row owned by this lane
  int bsel = (MODE == 2) ? ((z & 1) * 2 + (y0 & 1)) : 0;

  v8f acc[MROWS][NTILES];
#pragma unroll
  for (int my = 0; my < MROWS; ++my)
#pragma unroll
    for (int q = 0; q < NTILES; ++q) acc[my][q] = (v8f){};

  // Gather the A fragment for K-chunk r (both MROWS rows).
  auto gatherA = [&](int r, v16h a[MROWS]) {
#pragma unroll
    for (int my = 0; my < MROWS; ++my) {
      int y = y0 + my;
#pragma unroll
      for (int g = 0; g < 2; ++g) {
        int kk0 = r * 32 + h * 16 + (g ? 8 : 0);  // 8 consecutive K values
        int cbase = kk0 & (CIN - 1);
        bool valid;
        long pos = 0;
        if (MODE == 0) {
          int t = kk0 >> CSH;
          int dz = t / 9, rem = t - dz * 9, dy = rem / 3, dx = rem - dy * 3;
          int pz = z + dz - 1, py = y + dy - 1, px = x + dx - 1;
          valid = (t < 27) && ((unsigned)pz < (unsigned)Din) &&
                  ((unsigned)py < (unsigned)Hin) && ((unsigned)px < (unsigned)Win);
          pos = ((long)pz * Hin + py) * (long)Win + px;
        } else if (MODE == 1) {
          int t = kk0 >> CSH;
          int dz = (t >> 2) & 1, dy = (t >> 1) & 1, dx = t & 1;
          int pz = 2 * z + dz, py = 2 * y + dy, px = 2 * x + dx;
          valid = (x < Wout);
          pos = ((long)pz * Hin + py) * (long)Win + px;
        } else {
          int xp = kk0 >> CSH;
          valid = ((x & 1) == xp) && (x < Wout);
          pos = ((long)(z >> 1) * Hin + (y >> 1)) * (long)Win + (x >> 1);
        }
        v8h gv = {};
        if (valid) gv = *(const v8h*)(in + pos * inStride + inOff + cbase);
#pragma unroll
        for (int jj = 0; jj < 8; ++jj) a[my][g * 8 + jj] = gv[jj];
      }
    }
  };

  if constexpr (LDSB) {
    for (int r0 = 0; r0 < R; r0 += RB) {
      __syncthreads();  // previous slab fully consumed
      // Cooperative stage of RB chunks x NTILES into LDS (full slab; tail
      // over-read stays inside the guarded workspace).  Record e = 32B.
#pragma unroll
      for (int it = 0; it < (NTILES * RB * 32) / (32 * WVS); ++it) {
        int e = it * (32 * WVS) + tid;
        int l2 = e & 31;
        int rr = (e >> 5) & (RB - 1);
        int q = e >> 8;
        v16h t16 = *(const v16h*)(bsw + (size_t)(q * R + (r0 + rr)) * 512 + l2 * 16);
        *(v16h*)(lb + (size_t)e * 16) = t16;
      }
      __syncthreads();
#pragma unroll
      for (int rr = 0; rr < RB; ++rr) {
        if (r0 + rr >= R) break;  // uniform
        v16h a[MROWS];
        gatherA(r0 + rr, a);
#pragma unroll
        for (int q = 0; q < NTILES; ++q) {
          v16h b = *(const v16h*)(lb + (size_t)((q * RB + rr) * 32 + lane) * 16);
#pragma unroll
          for (int my = 0; my < MROWS; ++my)
            acc[my][q] = __builtin_amdgcn_wmma_f32_16x16x32_f16(
                false, a[my], false, b, (short)0, acc[my][q], false, false);
        }
      }
    }
  } else {
#pragma unroll
    for (int r = 0; r < R; ++r) {
      v16h a[MROWS];
      gatherA(r, a);
#pragma unroll
      for (int q = 0; q < NTILES; ++q) {
        v16h b = *(const v16h*)(bsw + (size_t)((q * BV + bsel) * R + r) * 512 + lane * 16);
#pragma unroll
        for (int my = 0; my < MROWS; ++my)
          acc[my][q] = __builtin_amdgcn_wmma_f32_16x16x32_f16(
              false, a[my], false, b, (short)0, acc[my][q], false, false);
      }
    }
  }

  // Epilogue: C/D layout -> lane holds (M = r8 + 8h, N = m).
  int xh = xb + (h << 3);
#pragma unroll
  for (int my = 0; my < MROWS; ++my) {
    int y = y0 + my;
    long rowbase = ((long)z * Hout + y) * (long)Wout;
    v8f mv;
    if (xh + 8 <= Wout) {
      mv = *(const v8f*)(mask + rowbase + xh);
    } else {
#pragma unroll
      for (int jj = 0; jj < 8; ++jj)
        mv[jj] = (xh + jj < Wout) ? mask[rowbase + xh + jj] : 0.f;
    }
#pragma unroll
    for (int q = 0; q < NTILES; ++q) {
      int ch = q * 16 + m;
#pragma unroll
      for (int r8 = 0; r8 < 8; ++r8) {
        int xo = xh + r8;
        if (xo >= Wout) continue;
        float v = acc[my][q][r8] * mv[r8];
        if (relu) v = fmaxf(v, 0.f);
        if (ch < coutValid) {
          if (outf) outf[(long)ch * outVol + rowbase + xo] = v;
          else      outh[(rowbase + xo) * outStride + outOff + ch] = (_Float16)v;
        }
      }
    }
  }
}

// ---------------------------------------------------------------------------
template <int MODE, int CIN, int R, int NTILES, int MROWS, int WVS>
static void launch_conv(hipStream_t stream, const _Float16* in, const _Float16* bsw,
                        const float* mask, _Float16* oh, float* of,
                        int Din, int Hin, int Win, int Dout, int Hout, int Wout,
                        int inS, int inO, int outS, int outO, int relu, int cv,
                        long ovol) {
  int tilesX = (Wout + 15) / 16;
  dim3 grid((unsigned)tilesX, (unsigned)(Hout / MROWS), (unsigned)(Dout / WVS));
  conv_wmma<MODE, CIN, R, NTILES, MROWS, WVS><<<grid, 32 * WVS, 0, stream>>>(
      in, bsw, mask, oh, of, Din, Hin, Win, Dout, Hout, Wout,
      inS, inO, outS, outO, relu, cv, ovol);
}

// ---------------------------------------------------------------------------
extern "C" void kernel_launch(void* const* d_in, const int* in_sizes, int n_in,
                              void* d_out, int out_size, void* d_ws, size_t ws_size,
                              hipStream_t stream) {
  const float* x     = (const float*)d_in[0];
  const float* occ   = (const float*)d_in[1];
  const float* w_in  = (const float*)d_in[2];
  const float* w_e0d = (const float*)d_in[3];
  const float* w_e0a = (const float*)d_in[4];
  const float* w_e0b = (const float*)d_in[5];
  const float* w_e1d = (const float*)d_in[6];
  const float* w_e1a = (const float*)d_in[7];
  const float* w_e1b = (const float*)d_in[8];
  const float* w_d0u = (const float*)d_in[9];
  const float* w_d0a = (const float*)d_in[10];
  const float* w_d0b = (const float*)d_in[11];
  const float* w_d1u = (const float*)d_in[12];
  const float* w_d1a = (const float*)d_in[13];
  const float* w_d1b = (const float*)d_in[14];
  const float* w_oa  = (const float*)d_in[15];
  const float* w_ob  = (const float*)d_in[16];

  const long V0 = 96L * 96 * 96, V1 = 48L * 48 * 48, V2 = 24L * 24 * 24;

  char* ws = (char*)d_ws;
  size_t off = 0;
  auto carve = [&](size_t bytes) -> char* {
    char* p = ws + off;
    off = (off + bytes + 255) & ~(size_t)255;
    return p;
  };

  _Float16* A_IN = (_Float16*)carve(V0 * 16 * 2);
  _Float16* CAT0 = (_Float16*)carve(V0 * 32 * 2);  // [dec1_up | skip0]
  _Float16* T0A  = (_Float16*)carve(V0 * 32 * 2);  // dec1_a out
  _Float16* CAT1 = (_Float16*)carve(V1 * 64 * 2);  // [dec0_up | skip1]
  _Float16* L1A  = (_Float16*)carve(V1 * 32 * 2);
  _Float16* L1B  = (_Float16*)carve(V1 * 32 * 2);
  _Float16* D1A  = (_Float16*)carve(V1 * 64 * 2);  // dec0_a out
  _Float16* L2A  = (_Float16*)carve(V2 * 64 * 2);
  _Float16* L2B  = (_Float16*)carve(V2 * 64 * 2);
  float* M1 = (float*)carve(V1 * 4);
  float* M2 = (float*)carve(V2 * 4);

  auto bc = [&](long e) { return (_Float16*)carve(e * 2); };
  _Float16* B_win = bc(14L * 512);
  _Float16* B_e0d = bc(2L * 4 * 512);
  _Float16* B_e0a = bc(2L * 27 * 512);
  _Float16* B_e0b = bc(2L * 27 * 512);
  _Float16* B_e1d = bc(4L * 8 * 512);
  _Float16* B_e1a = bc(4L * 54 * 512);
  _Float16* B_e1b = bc(4L * 54 * 512);
  _Float16* B_d0u = bc(2L * 4 * 4 * 512);
  _Float16* B_d0a = bc(4L * 54 * 512);
  _Float16* B_d0b = bc(2L * 54 * 512);
  _Float16* B_d1u = bc(1L * 4 * 2 * 512);
  _Float16* B_d1a = bc(2L * 27 * 512);
  _Float16* B_d1b = bc(1L * 27 * 512);
  _Float16* B_oa  = bc(14L * 512);
  _Float16* B_ob  = bc(14L * 512);
  (void)carve(32768);  // guard: LDS slab staging may over-read past last panel

  _Float16* T0B = A_IN;  // dec1_b out (A_IN dead after w_in layer)
  _Float16* T0C = CAT0;  // out_a out  (CAT0 dead after dec1_a)

  auto swz = [&](const float* w, _Float16* b, int actCin, int wCin, int Cout,
                 int T, int R, int Nt, int mode) {
    long total = (long)Nt * ((mode == 2) ? 4 : 1) * R * 512;
    swizzle_w<<<(unsigned)((total + 255) / 256), 256, 0, stream>>>(
        w, b, actCin, wCin, Cout, T, R, Nt, mode);
  };
  swz(w_in,  B_win, 16, 1,  16, 27, 14, 1, 0);
  swz(w_e0d, B_e0d, 16, 16, 32, 8,  4,  2, 1);
  swz(w_e0a, B_e0a, 32, 32, 32, 27, 27, 2, 0);
  swz(w_e0b, B_e0b, 32, 32, 32, 27, 27, 2, 0);
  swz(w_e1d, B_e1d, 32, 32, 64, 8,  8,  4, 1);
  swz(w_e1a, B_e1a, 64, 64, 64, 27, 54, 4, 0);
  swz(w_e1b, B_e1b, 64, 64, 64, 27, 54, 4, 0);
  swz(w_d0u, B_d0u, 64, 64, 32, 8,  4,  2, 2);
  swz(w_d0a, B_d0a, 64, 64, 64, 27, 54, 4, 0);
  swz(w_d0b, B_d0b, 64, 64, 32, 27, 54, 2, 0);
  swz(w_d1u, B_d1u, 32, 32, 16, 8,  2,  1, 2);
  swz(w_d1a, B_d1a, 32, 32, 32, 27, 27, 2, 0);
  swz(w_d1b, B_d1b, 32, 32, 16, 27, 27, 1, 0);
  swz(w_oa,  B_oa,  16, 16, 16, 27, 14, 1, 0);
  swz(w_ob,  B_ob,  16, 16, 5,  27, 14, 1, 0);

  prep_in<<<(unsigned)((V0 + 255) / 256), 256, 0, stream>>>(x, occ, A_IN, V0);
  pool2<<<(unsigned)((V1 + 255) / 256), 256, 0, stream>>>(occ, M1, 48, 48, 48);
  pool2<<<(unsigned)((V2 + 255) / 256), 256, 0, stream>>>(M1, M2, 24, 24, 24);

  // input block (skip0 -> CAT0 @ch16)
  launch_conv<0, 16, 14, 1, 2, 4>(stream, A_IN, B_win, occ, CAT0, nullptr,
                                  96, 96, 96, 96, 96, 96, 16, 0, 32, 16, 0, 16, 0);
  // encoder level 0
  launch_conv<1, 16, 4, 2, 2, 4>(stream, CAT0, B_e0d, M1, L1A, nullptr,
                                 96, 96, 96, 48, 48, 48, 32, 16, 32, 0, 0, 32, 0);
  launch_conv<0, 32, 27, 2, 2, 4>(stream, L1A, B_e0a, M1, L1B, nullptr,
                                  48, 48, 48, 48, 48, 48, 32, 0, 32, 0, 1, 32, 0);
  launch_conv<0, 32, 27, 2, 2, 4>(stream, L1B, B_e0b, M1, CAT1, nullptr,
                                  48, 48, 48, 48, 48, 48, 32, 0, 64, 32, 0, 32, 0);
  // encoder level 1
  launch_conv<1, 32, 8, 4, 2, 4>(stream, CAT1, B_e1d, M2, L2A, nullptr,
                                 48, 48, 48, 24, 24, 24, 64, 32, 64, 0, 0, 64, 0);
  launch_conv<0, 64, 54, 4, 2, 4>(stream, L2A, B_e1a, M2, L2B, nullptr,
                                  24, 24, 24, 24, 24, 24, 64, 0, 64, 0, 1, 64, 0);
  launch_conv<0, 64, 54, 4, 2, 4>(stream, L2B, B_e1b, M2, L2A, nullptr,
                                  24, 24, 24, 24, 24, 24, 64, 0, 64, 0, 0, 64, 0);
  // decoder stage 0
  launch_conv<2, 64, 4, 2, 1, 1>(stream, L2A, B_d0u, M1, CAT1, nullptr,
                                 24, 24, 24, 48, 48, 48, 64, 0, 64, 0, 0, 32, 0);
  launch_conv<0, 64, 54, 4, 2, 4>(stream, CAT1, B_d0a, M1, D1A, nullptr,
                                  48, 48, 48, 48, 48, 48, 64, 0, 64, 0, 1, 64, 0);
  launch_conv<0, 64, 54, 2, 2, 4>(stream, D1A, B_d0b, M1, L1A, nullptr,
                                  48, 48, 48, 48, 48, 48, 64, 0, 32, 0, 0, 32, 0);
  // decoder stage 1
  launch_conv<2, 32, 2, 1, 1, 1>(stream, L1A, B_d1u, occ, CAT0, nullptr,
                                 48, 48, 48, 96, 96, 96, 32, 0, 32, 0, 0, 16, 0);
  launch_conv<0, 32, 27, 2, 2, 4>(stream, CAT0, B_d1a, occ, T0A, nullptr,
                                  96, 96, 96, 96, 96, 96, 32, 0, 32, 0, 1, 32, 0);
  launch_conv<0, 32, 27, 1, 2, 4>(stream, T0A, B_d1b, occ, T0B, nullptr,
                                  96, 96, 96, 96, 96, 96, 32, 0, 16, 0, 0, 16, 0);
  // output block
  launch_conv<0, 16, 14, 1, 2, 4>(stream, T0B, B_oa, occ, T0C, nullptr,
                                  96, 96, 96, 96, 96, 96, 16, 0, 16, 0, 1, 16, 0);
  launch_conv<0, 16, 14, 1, 2, 4>(stream, T0C, B_ob, occ, nullptr, (float*)d_out,
                                  96, 96, 96, 96, 96, 96, 16, 0, 0, 0, 0, 5, V0);
}
